// AttentionFlowLayer_69569880260799
// MI455X (gfx1250) — matrix operand order repaired and checked by reading, not compiled, not run on hardware
//
#include <hip/hip_runtime.h>
#include <hip/hip_bf16.h>

typedef __attribute__((ext_vector_type(16))) __bf16 v16bf;
typedef __attribute__((ext_vector_type(8)))  float  v8f;

constexpr int kB = 64;     // batch
constexpr int kT = 2048;   // context length
constexpr int kJ = 128;    // query length
constexpr int kD = 200;    // feature dim
constexpr int KP = 224;    // K padded to multiple of 32 for bf16 WMMA
constexpr int US = 232;    // U_lds row stride (bf16 elems), padded vs banks
constexpr int PS = 136;    // P_lds row stride (bf16 elems), padded vs banks
constexpr int UTR = 208;   // transposed-U rows (covers 13 N-tiles of 16)
constexpr int UTS = 136;   // transposed-U row stride (j), padded vs banks
constexpr int TT = 128;    // t-rows per workgroup
constexpr int NW = TT / 16;        // 8 waves
constexpr int NTHR = NW * 32;      // 256 threads
constexpr int GW = 4 * kD;         // 800 floats per G row

// Split f32 into bf16 hi + bf16 lo so that hi + lo ~= x to ~2^-17 relative.
__device__ __forceinline__ void bf16_split(float x, __bf16& hi, __bf16& lo) {
    hi = (__bf16)x;
    lo = (__bf16)(x - (float)hi);
}

// ---------------------------------------------------------------------------
// Pass A: per (b, t-tile): S = (H.w_m) U^T + a[t] + c[j] + b ; softmax over j;
// rowmax -> ws ; U_tilde = P.U ; write G[...,0:600]
// Both GEMMs use bf16x3 (hi/lo split) WMMA for ~fp32 accuracy.
// ---------------------------------------------------------------------------
__global__ __launch_bounds__(NTHR)
void bidaf_passA(const float* __restrict__ H, const float* __restrict__ U,
                 const float* __restrict__ w, const float* __restrict__ bptr,
                 float* __restrict__ G, float* __restrict__ mrow)
{
    __shared__ __bf16 Uh [kJ * US];    // U row-major hi (stage-1 B frags)
    __shared__ __bf16 Ul [kJ * US];    // U row-major lo
    __shared__ __bf16 Uth[UTR * UTS];  // U transposed hi (stage-2 B frags)
    __shared__ __bf16 Utl[UTR * UTS];  // U transposed lo
    __shared__ __bf16 Ph [TT * PS];    // softmax probs hi
    __shared__ __bf16 Pl [TT * PS];    // softmax probs lo
    __shared__ float  wlds[3 * kD];
    __shared__ float  alds[TT];        // H.w_h per row
    __shared__ float  clds[kJ];        // U.w_u + b per col

    const int tid = threadIdx.x;
    const int b   = blockIdx.y;
    const int t0  = blockIdx.x * TT;
    const float bias = bptr[0];

    const float* Hb = H + (size_t)b * kT * kD;
    const float* Ub = U + (size_t)b * kJ * kD;

    for (int i = tid; i < 3 * kD; i += NTHR) wlds[i] = w[i];
    for (int i = tid; i < kJ * US; i += NTHR) {
        int j = i / US, d = i - j * US;
        float v = (d < kD) ? Ub[(size_t)j * kD + d] : 0.0f;
        __bf16 hi, lo;
        bf16_split(v, hi, lo);
        Uh[i] = hi;
        Ul[i] = lo;
        if (d < UTR) {                      // transposed copies for stage 2
            Uth[d * UTS + j] = hi;
            Utl[d * UTS + j] = lo;
        }
    }
    __syncthreads();

    for (int r = tid; r < TT; r += NTHR) {
        const float* h = Hb + (size_t)(t0 + r) * kD;
        float s = 0.0f;
        for (int d = 0; d < kD; ++d) s += h[d] * wlds[d];
        alds[r] = s;
    }
    for (int j = tid; j < kJ; j += NTHR) {
        const float* u = Ub + (size_t)j * kD;
        float s = 0.0f;
        for (int d = 0; d < kD; ++d) s += u[d] * wlds[kD + d];
        clds[j] = s + bias;
    }
    __syncthreads();

    const int wv   = tid >> 5;
    const int lane = tid & 31;
    const int half = lane >> 4;   // 0: lanes 0-15, 1: lanes 16-31
    const int l16  = lane & 15;

    // ---------------- Stage 1: S' = (H.w_m) x U^T, 16 rows x 128 cols / wave
    v8f acc[8] = {};
    const float* hrow = Hb + (size_t)(t0 + wv * 16 + l16) * kD;

    for (int k0 = 0; k0 < KP; k0 += 32) {
        __builtin_prefetch(hrow + k0 + 32, 0, 1);   // global_prefetch next chunk
        // A fragment (contiguous k order, same as B): lanes<16: K = k0+{0..15};
        // lanes>=16: K = k0+{16..31}; M = l16
        v16bf ah, al;
        const int kb0 = k0 + (half ? 16 : 0);
        const int kb1 = k0 + (half ? 24 : 8);
#pragma unroll
        for (int e = 0; e < 8; ++e) {
            int k = kb0 + e;
            float x = (k < kD) ? hrow[k] * wlds[2 * kD + k] : 0.0f;
            __bf16 hi, lo; bf16_split(x, hi, lo);
            ah[e] = hi; al[e] = lo;
        }
#pragma unroll
        for (int e = 0; e < 8; ++e) {
            int k = kb1 + e;
            float x = (k < kD) ? hrow[k] * wlds[2 * kD + k] : 0.0f;
            __bf16 hi, lo; bf16_split(x, hi, lo);
            ah[8 + e] = hi; al[8 + e] = lo;
        }
#pragma unroll
        for (int jt = 0; jt < 8; ++jt) {
            // B fragment (16-bit B 32x16): N = jt*16 + l16, K = k0 + half*16 + e
            const int uoff = (jt * 16 + l16) * US + k0 + half * 16;
            v16bf bh, bl;
#pragma unroll
            for (int e = 0; e < 16; ++e) { bh[e] = Uh[uoff + e]; bl[e] = Ul[uoff + e]; }
            acc[jt] = __builtin_amdgcn_wmma_f32_16x16x32_bf16(
                false, al, false, bh, (short)0, acc[jt], false, false);
            acc[jt] = __builtin_amdgcn_wmma_f32_16x16x32_bf16(
                false, ah, false, bl, (short)0, acc[jt], false, false);
            acc[jt] = __builtin_amdgcn_wmma_f32_16x16x32_bf16(
                false, ah, false, bh, (short)0, acc[jt], false, false);
        }
    }

    // ---------------- bias add + softmax over J per row; rowmax -> ws
#pragma unroll
    for (int r = 0; r < 8; ++r) {
        const int tl = wv * 16 + half * 8 + r;   // C/D layout: row = half*8 + reg
        const float a_t = alds[tl];
        float v[8];
        float mx = -__builtin_inff();
#pragma unroll
        for (int jt = 0; jt < 8; ++jt) {
            v[jt] = acc[jt][r] + a_t + clds[jt * 16 + l16];
            mx = fmaxf(mx, v[jt]);
        }
#pragma unroll
        for (int off = 1; off < 16; off <<= 1)
            mx = fmaxf(mx, __shfl_xor(mx, off, 32));
        float sum = 0.0f;
#pragma unroll
        for (int jt = 0; jt < 8; ++jt) { v[jt] = __expf(v[jt] - mx); sum += v[jt]; }
#pragma unroll
        for (int off = 1; off < 16; off <<= 1)
            sum += __shfl_xor(sum, off, 32);
        const float inv = 1.0f / sum;
#pragma unroll
        for (int jt = 0; jt < 8; ++jt) {
            __bf16 hi, lo; bf16_split(v[jt] * inv, hi, lo);
            Ph[tl * PS + jt * 16 + l16] = hi;
            Pl[tl * PS + jt * 16 + l16] = lo;
        }
        if (l16 == 0) mrow[(size_t)b * kT + t0 + tl] = mx;
    }
    __syncthreads();

    // ---------------- Stage 2: U_tilde = P x U (K = J = 128), 13 N-tiles
    for (int nt = 0; nt < 13; ++nt) {
        v8f acc2 = {};
        const int dcol = nt * 16 + l16;
#pragma unroll
        for (int j0 = 0; j0 < kJ; j0 += 32) {
            v16bf ah, al;  // A = P rows: M = l16, K = j pattern
            const int poff = (wv * 16 + l16) * PS;
            const int jb0 = j0 + (half ? 16 : 0);
            const int jb1 = j0 + (half ? 24 : 8);
#pragma unroll
            for (int e = 0; e < 8; ++e) { ah[e] = Ph[poff + jb0 + e]; al[e] = Pl[poff + jb0 + e]; }
#pragma unroll
            for (int e = 0; e < 8; ++e) { ah[8 + e] = Ph[poff + jb1 + e]; al[8 + e] = Pl[poff + jb1 + e]; }
            v16bf bh, bl;  // B = U^T: N = dcol, K = j contiguous (transposed copy)
            const int toff = dcol * UTS + j0 + half * 16;
#pragma unroll
            for (int e = 0; e < 16; ++e) { bh[e] = Uth[toff + e]; bl[e] = Utl[toff + e]; }
            acc2 = __builtin_amdgcn_wmma_f32_16x16x32_bf16(
                false, al, false, bh, (short)0, acc2, false, false);
            acc2 = __builtin_amdgcn_wmma_f32_16x16x32_bf16(
                false, ah, false, bl, (short)0, acc2, false, false);
            acc2 = __builtin_amdgcn_wmma_f32_16x16x32_bf16(
                false, ah, false, bh, (short)0, acc2, false, false);
        }
        if (dcol < kD) {
#pragma unroll
            for (int r = 0; r < 8; ++r) {
                const int tl = wv * 16 + half * 8 + r;
                const size_t t = (size_t)(t0 + tl);
                const float ut = acc2[r];
                const float h  = Hb[t * kD + dcol];
                float* g = G + ((size_t)b * kT + t) * GW;
                g[dcol]          = h;
                g[kD + dcol]     = ut;
                g[2 * kD + dcol] = h * ut;
            }
        }
    }
}

// ---------------------------------------------------------------------------
// Pass B: per batch: Q2C = softmax_t(rowmax) ; H_tilde = sum_t Q2C[t]*H[t,:]
// ---------------------------------------------------------------------------
__global__ __launch_bounds__(256)
void bidaf_passB(const float* __restrict__ H, const float* __restrict__ mrow,
                 float* __restrict__ q2c, float* __restrict__ htilde)
{
    __shared__ float qlds[kT];
    __shared__ float red[256];
    __shared__ float hpart[8][kD];

    const int b = blockIdx.x, tid = threadIdx.x;
    const float* mb = mrow + (size_t)b * kT;

    float mx = -__builtin_inff();
    for (int t = tid; t < kT; t += 256) mx = fmaxf(mx, mb[t]);
    red[tid] = mx; __syncthreads();
    for (int s = 128; s > 0; s >>= 1) {
        if (tid < s) red[tid] = fmaxf(red[tid], red[tid + s]);
        __syncthreads();
    }
    mx = red[0]; __syncthreads();

    float sum = 0.0f;
    for (int t = tid; t < kT; t += 256) {
        float e = __expf(mb[t] - mx);
        qlds[t] = e; sum += e;
    }
    red[tid] = sum; __syncthreads();
    for (int s = 128; s > 0; s >>= 1) {
        if (tid < s) red[tid] += red[tid + s];
        __syncthreads();
    }
    const float inv = 1.0f / red[0];
    for (int t = tid; t < kT; t += 256) {
        float q = qlds[t] * inv;
        qlds[t] = q;
        q2c[(size_t)b * kT + t] = q;
    }
    __syncthreads();

    const int wv = tid >> 5, lane = tid & 31;
    const float* Hb = H + (size_t)b * kT * kD;
    float accv[7] = {};
    for (int t = wv * 256; t < (wv + 1) * 256; ++t) {
        const float q = qlds[t];
        const float* h = Hb + (size_t)t * kD;
#pragma unroll
        for (int k = 0; k < 7; ++k) {
            int dd = lane + 32 * k;
            if (dd < kD) accv[k] += q * h[dd];
        }
    }
#pragma unroll
    for (int k = 0; k < 7; ++k) {
        int dd = lane + 32 * k;
        if (dd < kD) hpart[wv][dd] = accv[k];
    }
    __syncthreads();
    if (tid < kD) {
        float s2 = 0.0f;
        for (int i = 0; i < 8; ++i) s2 += hpart[i][tid];
        htilde[(size_t)b * kD + tid] = s2;
    }
}

// ---------------------------------------------------------------------------
// Pass C: G[..., 600:800] = H * H_tilde
// ---------------------------------------------------------------------------
__global__ __launch_bounds__(256)
void bidaf_passC(const float* __restrict__ H, const float* __restrict__ htilde,
                 float* __restrict__ G)
{
    const size_t idx = (size_t)blockIdx.x * 256 + threadIdx.x;
    const size_t total = (size_t)kB * kT * kD;
    if (idx >= total) return;
    const int d = (int)(idx % kD);
    const size_t bt = idx / kD;
    const int b = (int)(bt / kT);
    const float h = H[idx];
    G[bt * GW + 3 * kD + d] = h * htilde[(size_t)b * kD + d];
}

extern "C" void kernel_launch(void* const* d_in, const int* in_sizes, int n_in,
                              void* d_out, int out_size, void* d_ws, size_t ws_size,
                              hipStream_t stream)
{
    const float* H  = (const float*)d_in[0];
    const float* U  = (const float*)d_in[1];
    const float* w  = (const float*)d_in[2];
    const float* bp = (const float*)d_in[3];

    float* G    = (float*)d_out;
    float* q2c  = G + (size_t)kB * kT * GW;     // second output, flat after G
    float* mrow = (float*)d_ws;                 // B*T rowmax
    float* htil = mrow + (size_t)kB * kT;       // B*D H_tilde

    dim3 gA(kT / TT, kB);
    bidaf_passA<<<gA, NTHR, 0, stream>>>(H, U, w, bp, G, mrow);
    bidaf_passB<<<kB, 256, 0, stream>>>(H, mrow, q2c, htil);
    const size_t totalC = (size_t)kB * kT * kD;
    bidaf_passC<<<(unsigned)((totalC + 255) / 256), 256, 0, stream>>>(H, htil, G);
}
